// Linear_38208029066069
// MI455X (gfx1250) — matrix operand order repaired and checked by reading, not compiled, not run on hardware
//
#include <hip/hip_runtime.h>

typedef __attribute__((ext_vector_type(16))) __bf16          v16bf;
typedef __attribute__((ext_vector_type(2)))  __bf16          bf16x2;
typedef __attribute__((ext_vector_type(8)))  float           v8f;
typedef __attribute__((ext_vector_type(4)))  float           f32x4;
typedef __attribute__((ext_vector_type(4)))  int             i32x4;
typedef __attribute__((ext_vector_type(2)))  unsigned int    u32x2;
typedef __attribute__((ext_vector_type(4)))  unsigned int    u32x4;
typedef __attribute__((ext_vector_type(8)))  unsigned short  u16x8;
typedef __attribute__((ext_vector_type(16))) unsigned short  u16x16;

#define M_DIM   8192
#define N_DIM   4096
#define K_DIM   4096
#define GROUP   128
#define NGROUPS (K_DIM / GROUP)

#define BM 128
#define BN 128
#define BK 64
#define LDSS 72          // BK + 8 pad (shorts); row = 144 B, 16B aligned
#define TILE (BM * LDSS) // shorts per buffer

// pack two floats -> packed bf16 pair (v_cvt_pk_bf16_f32)
__device__ __forceinline__ unsigned int pk_bf16(float lo, float hi) {
    bf16x2 v;
    v.x = (__bf16)lo;
    v.y = (__bf16)hi;
    return __builtin_bit_cast(unsigned int, v);
}

__device__ __forceinline__ v16bf frag16(u16x8 lo, u16x8 hi) {
    u16x16 t = __builtin_shufflevector(lo, hi, 0,1,2,3,4,5,6,7,8,9,10,11,12,13,14,15);
    return __builtin_bit_cast(v16bf, t);
}

// stage one BK-chunk: A (f32->bf16) and W (nibble dequant->bf16) into LDS buffer
__device__ __forceinline__ void stage_chunk(
    const float* __restrict__ Abase, const int* __restrict__ Qbase,
    const float* __restrict__ Srow,  const float* __restrict__ Zrow,
    unsigned short* sArow, unsigned short* sWrow, int k0)
{
    const f32x4* ap = (const f32x4*)(Abase + k0);
    const int    g   = k0 / GROUP;
    const float  sc  = Srow[g];
    const float  nzs = -Zrow[g] * sc;        // (w - z)*s = fma(w, s, -z*s)
    const i32x4* qp  = (const i32x4*)(Qbase + (k0 >> 1));

#pragma unroll
    for (int i = 0; i < 8; ++i) {            // 32 floats of A per thread
        f32x4 v = ap[i];
        u32x2 b;
        b.x = pk_bf16(v.x, v.y);
        b.y = pk_bf16(v.z, v.w);
        *(u32x2*)(sArow + i * 4) = b;
    }
#pragma unroll
    for (int i = 0; i < 4; ++i) {            // 16 qweight ints per thread
        i32x4 q = qp[i];
        u32x4 w;
        w.x = pk_bf16(__builtin_fmaf((float)( q.x       & 0xF), sc, nzs),
                      __builtin_fmaf((float)((q.x >> 4) & 0xF), sc, nzs));
        w.y = pk_bf16(__builtin_fmaf((float)( q.y       & 0xF), sc, nzs),
                      __builtin_fmaf((float)((q.y >> 4) & 0xF), sc, nzs));
        w.z = pk_bf16(__builtin_fmaf((float)( q.z       & 0xF), sc, nzs),
                      __builtin_fmaf((float)((q.z >> 4) & 0xF), sc, nzs));
        w.w = pk_bf16(__builtin_fmaf((float)( q.w       & 0xF), sc, nzs),
                      __builtin_fmaf((float)((q.w >> 4) & 0xF), sc, nzs));
        *(u32x4*)(sWrow + i * 8) = w;
    }
}

// two K=32 WMMA steps over one staged BK=64 chunk
__device__ __forceinline__ void compute_chunk(
    const unsigned short* sAfrag, const unsigned short* sWfrag, v8f acc[2][4])
{
#pragma unroll
    for (int kk = 0; kk < BK; kk += 32) {
        v16bf af[2];
#pragma unroll
        for (int mi = 0; mi < 2; ++mi) {
            // A 16x32 bf16: lane<16 holds K {0..7,16..23}, lane>=16 holds {8..15,24..31}
            const unsigned short* p = sAfrag + mi * 16 * LDSS + kk;
            u16x8 lo = *(const u16x8*)(p);
            u16x8 hi = *(const u16x8*)(p + 16);
            af[mi] = frag16(lo, hi);
        }
        v16bf bfr[4];
#pragma unroll
        for (int ni = 0; ni < 4; ++ni) {
            // B 32x16 bf16: lane<16 holds K {0..7,16..23} of col N, lane>=16 holds {8..15,24..31}
            const unsigned short* p = sWfrag + ni * 16 * LDSS + kk;
            u16x8 lo = *(const u16x8*)(p);
            u16x8 hi = *(const u16x8*)(p + 16);
            bfr[ni] = frag16(lo, hi);
        }
#pragma unroll
        for (int mi = 0; mi < 2; ++mi)
#pragma unroll
            for (int ni = 0; ni < 4; ++ni)
                acc[mi][ni] = __builtin_amdgcn_wmma_f32_16x16x32_bf16(
                    false, af[mi], false, bfr[ni],
                    (short)0, acc[mi][ni], false, false);
    }
}

__global__ __launch_bounds__(256, 2) void gptq_wmma_gemm(
    const float* __restrict__ A,        // [M, K] f32
    const int*   __restrict__ qweight,  // [N, K/2] int32 (1 byte / entry, 2 nibbles)
    const float* __restrict__ scales,   // [N, NGROUPS]
    const float* __restrict__ zeros,    // [N, NGROUPS]
    float*       __restrict__ C)        // [M, N] f32
{
    // double-buffered tiles: 4 x 18 KB = 72 KB
    __shared__ unsigned short sA[2 * TILE];
    __shared__ unsigned short sW[2 * TILE];

    const int tid   = threadIdx.x;
    const int lane  = tid & 31;
    const int wave  = tid >> 5;
    const int waveM = wave >> 1;   // 0..3  -> 32 rows each
    const int waveN = wave & 1;    // 0..1  -> 64 cols each

    const int blockN0 = blockIdx.x * BN;
    const int blockM0 = blockIdx.y * BM;

    v8f acc[2][4];
#pragma unroll
    for (int mi = 0; mi < 2; ++mi)
#pragma unroll
        for (int ni = 0; ni < 4; ++ni)
            acc[mi][ni] = (v8f)0.0f;

    // ---- staging assignments: 2 threads per row, each covers half the BK chunk
    const int row  = tid >> 1;           // 0..127 (both A row and W row)
    const int half = tid & 1;            // 0/1

    const float* Abase = A + (size_t)(blockM0 + row) * K_DIM + half * 32;
    const int*   Qbase = qweight + (size_t)(blockN0 + row) * (K_DIM / 2) + half * 16;
    const float* Srow  = scales  + (size_t)(blockN0 + row) * NGROUPS;
    const float* Zrow  = zeros   + (size_t)(blockN0 + row) * NGROUPS;

    unsigned short* sArow0 = sA + row * LDSS + half * 32;
    unsigned short* sWrow0 = sW + row * LDSS + half * 32;
    unsigned short* sArow1 = sArow0 + TILE;
    unsigned short* sWrow1 = sWrow0 + TILE;

    // ---- fragment addressing (ISA 7.12.2 layouts)
    const int hsel = (lane >> 4);
    const int l15  = lane & 15;
    const unsigned short* sAfrag0 = sA + (waveM * 32 + l15) * LDSS + hsel * 8;
    const unsigned short* sWfrag0 = sW + (waveN * 64 + l15) * LDSS + hsel * 8;
    const unsigned short* sAfrag1 = sAfrag0 + TILE;
    const unsigned short* sWfrag1 = sWfrag0 + TILE;

    // prologue: stage chunk 0 into buffer 0
    stage_chunk(Abase, Qbase, Srow, Zrow, sArow0, sWrow0, 0);
    __syncthreads();

    // main loop, unrolled x2 over the two buffers (64 chunks total)
    for (int k0 = 0; k0 < K_DIM; k0 += 2 * BK) {
        // stage k0+BK into buf1 while computing k0 from buf0
        if (k0 + BK < K_DIM) {
            __builtin_prefetch(Abase + k0 + 2 * BK, 0, 0);
            __builtin_prefetch(Qbase + ((k0 + 2 * BK) >> 1), 0, 0);
            stage_chunk(Abase, Qbase, Srow, Zrow, sArow1, sWrow1, k0 + BK);
        }
        compute_chunk(sAfrag0, sWfrag0, acc);
        __syncthreads();

        // stage k0+2*BK into buf0 while computing k0+BK from buf1
        if (k0 + 2 * BK < K_DIM) {
            __builtin_prefetch(Abase + k0 + 3 * BK, 0, 0);
            __builtin_prefetch(Qbase + ((k0 + 3 * BK) >> 1), 0, 0);
            stage_chunk(Abase, Qbase, Srow, Zrow, sArow0, sWrow0, k0 + 2 * BK);
        }
        compute_chunk(sAfrag1, sWfrag1, acc);
        __syncthreads();
    }

    // ---- epilogue: C/D layout -> row = v + 8*(lane>>4), col = lane&15
#pragma unroll
    for (int mi = 0; mi < 2; ++mi) {
#pragma unroll
        for (int ni = 0; ni < 4; ++ni) {
            const int m0 = blockM0 + waveM * 32 + mi * 16 + hsel * 8;
            const int n  = blockN0 + waveN * 64 + ni * 16 + l15;
            float* out = C + (size_t)m0 * N_DIM + n;
#pragma unroll
            for (int v = 0; v < 8; ++v)
                out[(size_t)v * N_DIM] = acc[mi][ni][v];
        }
    }
}

extern "C" void kernel_launch(void* const* d_in, const int* in_sizes, int n_in,
                              void* d_out, int out_size, void* d_ws, size_t ws_size,
                              hipStream_t stream) {
    const float* A       = (const float*)d_in[0];
    const int*   qweight = (const int*)d_in[1];
    const float* scales  = (const float*)d_in[2];
    const float* zeros   = (const float*)d_in[3];
    float*       C       = (float*)d_out;

    dim3 grid(N_DIM / BN, M_DIM / BM);   // (32, 64)
    dim3 block(256);
    gptq_wmma_gemm<<<grid, block, 0, stream>>>(A, qweight, scales, zeros, C);
}
